// DWReg2DDecode3D_30322469110339
// MI455X (gfx1250) — matrix operand order repaired and checked by reading, not compiled, not run on hardware
//
#include <hip/hip_runtime.h>

typedef _Float16 half_t;
typedef __attribute__((ext_vector_type(8)))  _Float16 v8h;
typedef __attribute__((ext_vector_type(16))) _Float16 v16h;
typedef __attribute__((ext_vector_type(8)))  float    v8f;

#define S_NB 9

// ---------------- zero ----------------
__global__ void k_zero(float* p, long n) {
    long i = (long)blockIdx.x * blockDim.x + threadIdx.x;
    if (i < n) p[i] = 0.0f;
}

// ---------------- grid sample (bilinear) ----------------
// xp[b,p,c] = bilinear(feat[b,c,:,:], uvc[b,p])
__global__ void k_gridsample(const float* __restrict__ uv, const float* __restrict__ feat,
                             float* __restrict__ xp, int B, int P, int C, int H, int W) {
    long i = (long)blockIdx.x * blockDim.x + threadIdx.x;
    long total = (long)B * P * C;
    if (i >= total) return;
    int c = (int)(i % C);
    int p = (int)((i / C) % P);
    int b = (int)(i / ((long)C * P));
    float u = uv[((long)b * P + p) * 2 + 0];
    float v = uv[((long)b * P + p) * 2 + 1];
    float ux = fminf(fmaxf((u - 0.5f) * 2.0f, -1.0f), 1.0f);
    float vy = fminf(fmaxf((v - 0.5f) * 2.0f, -1.0f), 1.0f);
    float x = (ux + 1.0f) * 0.5f * (float)(W - 1);
    float y = (vy + 1.0f) * 0.5f * (float)(H - 1);
    float x0 = floorf(x), y0 = floorf(y);
    float wx = x - x0, wy = y - y0;
    int x0i = min(max((int)x0, 0), W - 1);
    int x1i = min(max((int)x0 + 1, 0), W - 1);
    int y0i = min(max((int)y0, 0), H - 1);
    int y1i = min(max((int)y0 + 1, 0), H - 1);
    const float* fb = feat + ((long)b * C + c) * H * W;
    float f00 = fb[y0i * W + x0i];
    float f01 = fb[y0i * W + x1i];
    float f10 = fb[y1i * W + x0i];
    float f11 = fb[y1i * W + x1i];
    xp[i] = f00 * (1.0f - wy) * (1.0f - wx) + f01 * (1.0f - wy) * wx
          + f10 * wy * (1.0f - wx) + f11 * wy * wx;
}

// ---------------- upsample: x4[b,v,c] = sum_p up[v,p] * xp[b,p,c] ----------------
__global__ void k_upsample(const float* __restrict__ up, const float* __restrict__ xp,
                           float* __restrict__ x4, int B, int Vn, int P, int C) {
    long i = (long)blockIdx.x * blockDim.x + threadIdx.x;
    long total = (long)B * Vn * C;
    if (i >= total) return;
    int c = (int)(i % C);
    int v = (int)((i / C) % Vn);
    int b = (int)(i / ((long)C * Vn));
    float s = 0.0f;
    for (int p = 0; p < P; ++p)
        s += up[v * P + p] * xp[((long)b * P + p) * C + c];
    x4[i] = s;
}

// ---------------- sparse pool: out[b,row[nz],c] += x[b,col[nz],c]*val[nz] ----------------
__global__ void k_pool(const float* __restrict__ x, const int* __restrict__ row,
                       const int* __restrict__ col, const float* __restrict__ val,
                       float* __restrict__ out, int B, int nnz, int C, int Nin, int Nout) {
    long i = (long)blockIdx.x * blockDim.x + threadIdx.x;
    long total = (long)B * nnz * C;
    if (i >= total) return;
    int c  = (int)(i % C);
    int nz = (int)((i / C) % nnz);
    int b  = (int)(i / ((long)C * nnz));
    float g = x[((long)b * Nin + col[nz]) * C + c] * val[nz];
    atomicAdd(&out[((long)b * Nout + row[nz]) * C + c], g);
}

// ------ depthwise (gather 9 neighbors, per-channel weights), store f16 for WMMA A ------
__global__ void k_depthwise(const float* __restrict__ x, const int* __restrict__ sidx,
                            const float* __restrict__ wd, half_t* __restrict__ dw,
                            int B, int N, int C) {
    long i = (long)blockIdx.x * blockDim.x + threadIdx.x;
    long total = (long)B * N * C;
    if (i >= total) return;
    int c = (int)(i % C);
    int n = (int)((i / C) % N);
    int b = (int)(i / ((long)C * N));
    const int* sp = sidx + (long)n * S_NB;
    float acc = 0.0f;
    #pragma unroll
    for (int s = 0; s < S_NB; ++s)
        acc += x[((long)b * N + sp[s]) * C + c] * wd[s * C + c];
    dw[i] = (half_t)acc;
}

// ---------------- wp (Ci x Co, f32) -> wpT (Co x Ci, f16) ----------------
__global__ void k_wpT(const float* __restrict__ wp, half_t* __restrict__ wpT, int Ci, int Co) {
    long i = (long)blockIdx.x * blockDim.x + threadIdx.x;
    long total = (long)Ci * Co;
    if (i >= total) return;
    int ci = (int)(i % Ci);
    int co = (int)(i / Ci);
    wpT[(long)co * Ci + ci] = (half_t)wp[(long)ci * Co + co];
}

// ---------------- pointwise GEMM via v_wmma_f32_16x16x32_f16 ----------------
// One wave computes a 16(node) x 64(outchan) tile with FOUR independent
// accumulators: per k-step, one A fragment load feeds 4 dependency-free WMMAs
// (A traffic /4, no serial C-chain on the XDL pipe). K-loop over Cin in 32s.
// A layout (16x32 f16): lg=lane>>4, m=lane&15; elems 0..15 = K kb+lg*16+e
//                       -> two contiguous 16B loads.
// B layout (32x16 f16): n = lane&15; elems 0..15 = K kb+lg*16+e from Co-major wpT
//                       -> one contiguous 32B load per 16-wide cout sub-tile.
// C/D (16x16 f32): elem r -> M = r + lg*8, N = lane&15.
__global__ void k_pointwise_wmma(const half_t* __restrict__ dw, const half_t* __restrict__ wpT,
                                 const float* __restrict__ bias, float* __restrict__ out,
                                 int N, int Cin, int Cout, int do_relu) {
    int wave = threadIdx.x >> 5;
    int lane = threadIdx.x & 31;
    int tile = blockIdx.x * 8 + wave;   // grid exact: B*(N/16)*(Cout/64)/8 blocks
    int ctiles = Cout >> 6;             // 64-wide out-channel tiles
    int ntiles = N >> 4;
    int ct = tile % ctiles;
    int nt = (tile / ctiles) % ntiles;
    int b  = tile / (ctiles * ntiles);
    int lg = lane >> 4;
    int lr = lane & 15;
    const half_t* arow  = dw  + ((long)b * N + nt * 16 + lr) * Cin;
    const half_t* brow0 = wpT + (long)(ct * 64 + lr) * Cin;      // j-th sub-tile: +j*16*Cin
    long bstep = (long)16 * Cin;

    v8f acc0 = {0.f, 0.f, 0.f, 0.f, 0.f, 0.f, 0.f, 0.f};
    v8f acc1 = acc0, acc2 = acc0, acc3 = acc0;

    for (int kb = 0; kb < Cin; kb += 32) {
        v8h alo = *reinterpret_cast<const v8h*>(arow + kb + lg * 16);
        v8h ahi = *reinterpret_cast<const v8h*>(arow + kb + lg * 16 + 8);
        v16h a  = __builtin_shufflevector(alo, ahi,
                                          0, 1, 2, 3, 4, 5, 6, 7,
                                          8, 9, 10, 11, 12, 13, 14, 15);
        long boff = kb + lg * 16;
        v16h b0 = *reinterpret_cast<const v16h*>(brow0 + boff);
        v16h b1 = *reinterpret_cast<const v16h*>(brow0 + bstep + boff);
        v16h b2 = *reinterpret_cast<const v16h*>(brow0 + 2 * bstep + boff);
        v16h b3 = *reinterpret_cast<const v16h*>(brow0 + 3 * bstep + boff);
        acc0 = __builtin_amdgcn_wmma_f32_16x16x32_f16(false, a, false, b0, (short)0, acc0, false, false);
        acc1 = __builtin_amdgcn_wmma_f32_16x16x32_f16(false, a, false, b1, (short)0, acc1, false, false);
        acc2 = __builtin_amdgcn_wmma_f32_16x16x32_f16(false, a, false, b2, (short)0, acc2, false, false);
        acc3 = __builtin_amdgcn_wmma_f32_16x16x32_f16(false, a, false, b3, (short)0, acc3, false, false);
    }

    long obase = ((long)b * N + nt * 16) * Cout + ct * 64 + lr;
    v8f accs[4] = {acc0, acc1, acc2, acc3};
    #pragma unroll
    for (int j = 0; j < 4; ++j) {
        float bv = bias[ct * 64 + j * 16 + lr];
        #pragma unroll
        for (int r = 0; r < 8; ++r) {
            float v = accs[j][r] + bv;
            if (do_relu) v = fmaxf(v, 0.0f);
            out[obase + (long)(r + lg * 8) * Cout + j * 16] = v;
        }
    }
}

// ---------------- head: (8192 nodes, 64ch) x (64x3) + bias, no relu ----------------
__global__ void k_head(const half_t* __restrict__ dwh, const float* __restrict__ wph,
                       const float* __restrict__ bh, float* __restrict__ out,
                       int B, int N, int Ci) {
    long i = (long)blockIdx.x * blockDim.x + threadIdx.x;
    long total = (long)B * N;
    if (i >= total) return;
    const half_t* row = dwh + i * Ci;
    float a0 = bh[0], a1 = bh[1], a2 = bh[2];
    for (int c = 0; c < Ci; ++c) {
        float x = (float)row[c];
        a0 += x * wph[c * 3 + 0];
        a1 += x * wph[c * 3 + 1];
        a2 += x * wph[c * 3 + 2];
    }
    out[i * 3 + 0] = a0;
    out[i * 3 + 1] = a1;
    out[i * 3 + 2] = a2;
}

// =======================================================================
extern "C" void kernel_launch(void* const* d_in, const int* in_sizes, int n_in,
                              void* d_out, int out_size, void* d_ws, size_t ws_size,
                              hipStream_t stream) {
    (void)in_sizes; (void)n_in; (void)out_size; (void)ws_size;

    const int B = 8, P = 21, C0 = 512, H = 32, W = 32;
    const int V0 = 8192, V1 = 4096, V2 = 2048, V3 = 1024, V4 = 512;

    // ---- inputs (setup_inputs dict order) ----
    const float* uv   = (const float*)d_in[0];
    const float* feat = (const float*)d_in[1];
    const float* up   = (const float*)d_in[2];
    const int*   s0   = (const int*)d_in[3];
    const int*   s1   = (const int*)d_in[4];
    const int*   s2   = (const int*)d_in[5];
    const int*   s3   = (const int*)d_in[6];
    const int*   rows[4] = {(const int*)d_in[7],  (const int*)d_in[10], (const int*)d_in[13], (const int*)d_in[16]};
    const int*   cols[4] = {(const int*)d_in[8],  (const int*)d_in[11], (const int*)d_in[14], (const int*)d_in[17]};
    const float* vals[4] = {(const float*)d_in[9], (const float*)d_in[12], (const float*)d_in[15], (const float*)d_in[18]};
    const float* wd0 = (const float*)d_in[19]; const float* wp0 = (const float*)d_in[20]; const float* b0 = (const float*)d_in[21];
    const float* wd1 = (const float*)d_in[22]; const float* wp1 = (const float*)d_in[23]; const float* b1 = (const float*)d_in[24];
    const float* wd2 = (const float*)d_in[25]; const float* wp2 = (const float*)d_in[26]; const float* b2 = (const float*)d_in[27];
    const float* wd3 = (const float*)d_in[28]; const float* wp3 = (const float*)d_in[29]; const float* b3 = (const float*)d_in[30];
    const float* wdh = (const float*)d_in[31]; const float* wph = (const float*)d_in[32]; const float* bh = (const float*)d_in[33];
    float* outp = (float*)d_out;

    // ---- workspace layout (3 ping-pong slots + f16 weights + sample buf) ----
    char* ws = (char*)d_ws;
    const size_t SLOT_A = 0;                         // activations fp32, max 16.8 MB
    const size_t SLOT_B = SLOT_A + (size_t)16777216; // pooled fp32,     max 33.6 MB
    const size_t SLOT_C = SLOT_B + (size_t)33554432; // depthwise f16,   max 16.8 MB
    const size_t WPT0   = SLOT_C + (size_t)16777216;
    const size_t WPT1   = WPT0 + (size_t)512 * 512 * 2;
    const size_t WPT2   = WPT1 + (size_t)512 * 256 * 2;
    const size_t WPT3   = WPT2 + (size_t)256 * 128 * 2;
    const size_t XPOFF  = WPT3 + (size_t)128 * 64 * 2;
    float*  slotA = (float*)(ws + SLOT_A);
    float*  slotB = (float*)(ws + SLOT_B);
    half_t* slotC = (half_t*)(ws + SLOT_C);
    half_t* wpT[4] = {(half_t*)(ws + WPT0), (half_t*)(ws + WPT1),
                      (half_t*)(ws + WPT2), (half_t*)(ws + WPT3)};
    float* xp = (float*)(ws + XPOFF);

    auto blocks = [](long n) { return (unsigned)((n + 255) / 256); };

    // ---- 1. grid sample + upsample -> slotA = x4 (B, 512, 512) ----
    long ngs = (long)B * P * C0;
    k_gridsample<<<blocks(ngs), 256, 0, stream>>>(uv, feat, xp, B, P, C0, H, W);
    long nup = (long)B * V4 * C0;
    k_upsample<<<blocks(nup), 256, 0, stream>>>(up, xp, slotA, B, V4, P, C0);

    // ---- 2. f16 transposed pointwise weights ----
    k_wpT<<<blocks((long)512 * 512), 256, 0, stream>>>(wp0, wpT[0], 512, 512);
    k_wpT<<<blocks((long)512 * 256), 256, 0, stream>>>(wp1, wpT[1], 512, 256);
    k_wpT<<<blocks((long)256 * 128), 256, 0, stream>>>(wp2, wpT[2], 256, 128);
    k_wpT<<<blocks((long)128 * 64),  256, 0, stream>>>(wp3, wpT[3], 128, 64);

    // ---- 3. four {pool, depthwise, wmma-pointwise} stages ----
    struct Stage {
        int Nin, Cin, Nout, Cout, nnz, pool_idx;
        const int* sidx; const float* wd; const float* bias; int wpt;
    };
    Stage st[4] = {
        {V4, 512, V3, 512, 3 * V3, 3, s3, wd0, b0, 0},
        {V3, 512, V2, 256, 3 * V2, 2, s2, wd1, b1, 1},
        {V2, 256, V1, 128, 3 * V1, 1, s1, wd2, b2, 2},
        {V1, 128, V0,  64, 3 * V0, 0, s0, wd3, b3, 3},
    };
    for (int i = 0; i < 4; ++i) {
        const Stage& s = st[i];
        long npool_out = (long)B * s.Nout * s.Cin;
        k_zero<<<blocks(npool_out), 256, 0, stream>>>(slotB, npool_out);
        long npool = (long)B * s.nnz * s.Cin;
        k_pool<<<blocks(npool), 256, 0, stream>>>(slotA, rows[s.pool_idx], cols[s.pool_idx],
                                                  vals[s.pool_idx], slotB,
                                                  B, s.nnz, s.Cin, s.Nin, s.Nout);
        long ndw = (long)B * s.Nout * s.Cin;
        k_depthwise<<<blocks(ndw), 256, 0, stream>>>(slotB, s.sidx, s.wd, slotC,
                                                     B, s.Nout, s.Cin);
        int tiles = B * (s.Nout / 16) * (s.Cout / 64);     // 4096 every stage
        k_pointwise_wmma<<<tiles / 8, 256, 0, stream>>>(slotC, wpT[s.wpt], s.bias, slotA,
                                                        s.Nout, s.Cin, s.Cout, 1);
    }

    // ---- 4. head: depthwise on (8192, 64) then 64x3 dense ----
    long ndwh = (long)B * V0 * 64;
    k_depthwise<<<blocks(ndwh), 256, 0, stream>>>(slotA, s0, wdh, slotC, B, V0, 64);
    k_head<<<blocks((long)B * V0), 256, 0, stream>>>(slotC, wph, bh, outp, B, V0, 64);
}
